// HierarchicalLabelAttention_38062000177794
// MI455X (gfx1250) — matrix operand order repaired and checked by reading, not compiled, not run on hardware
//
#include <hip/hip_runtime.h>

// ---------------------------------------------------------------------------
// Types for CDNA5 WMMA (wave32, 16x16x32 bf16 -> f32)
// ---------------------------------------------------------------------------
typedef __attribute__((ext_vector_type(16))) __bf16        v16bf;
typedef __attribute__((ext_vector_type(2)))  __bf16        v2bf;
typedef __attribute__((ext_vector_type(8)))  float         v8f;
typedef __attribute__((ext_vector_type(4)))  unsigned int  v4u;

union FragU { v4u u[2]; v16bf v; };

// round-half-up bf16 (2 VALU ops)
__device__ __forceinline__ unsigned short f2bf(float f) {
    unsigned int u = __builtin_bit_cast(unsigned int, f);
    return (unsigned short)((u + 0x8000u) >> 16);
}

// pack two floats -> bf16x2 in one dword: 2 adds + 1 v_perm_b32
__device__ __forceinline__ unsigned int pack2bf(float a, float b) {
#if defined(__has_builtin) && __has_builtin(__builtin_amdgcn_cvt_pk_bf16_f32)
    v2bf r = __builtin_amdgcn_cvt_pk_bf16_f32(a, b);
    return __builtin_bit_cast(unsigned int, r);
#else
    unsigned int ua = __builtin_bit_cast(unsigned int, a) + 0x8000u;
    unsigned int ub = __builtin_bit_cast(unsigned int, b) + 0x8000u;
    // dst.b0=ua.b2, b1=ua.b3 (low16 = bf16(a)); b2=ub.b2, b3=ub.b3 (high16 = bf16(b))
    return __builtin_amdgcn_perm(ua, ub, 0x03020706u);
#endif
}

#define TM 128
#define TN 128
#define TK 32
#define LDS_S 40   // padded row stride (ushorts): 80B -> conflict-free b128 frag reads

// ---------------------------------------------------------------------------
// Tiled GEMM with register-prefetch pipeline:
//   C[b] = A[b] (MxK row-major) * Bmat[b] (KxN)  [+bias[n]] [*mul[m,n]]
//   bTransposed==1 : Bmat is [K,N] (n contiguous) -> transpose while staging
//   bTransposed==0 : Bmat is Bt=[N,K] (k contiguous) -> direct staging
// f32 globals, converted to bf16 on the way into LDS. blockIdx.z = batch.
// ---------------------------------------------------------------------------
__global__ __launch_bounds__(256)
void hla_wmma_gemm(const float* __restrict__ A, const float* __restrict__ Bm,
                   float* __restrict__ C,
                   int M, int N, int K, int lda, int ldb, int ldc,
                   long aBatch, long bBatch, long cBatch,
                   const float* __restrict__ bias,
                   const float* __restrict__ mul, int ldmul,
                   int bTransposed)
{
    __shared__ __align__(16) unsigned short As[TM * LDS_S];
    __shared__ __align__(16) unsigned short Bs[TN * LDS_S];

    const int tid     = threadIdx.x;
    const int lane    = tid & 31;
    const int wave    = tid >> 5;        // 0..7
    const int wm      = wave & 1;        // 64-row half
    const int wn      = wave >> 1;       // 32-col quarter
    const int laneRow = lane & 15;
    const int hiHalf  = lane >> 4;       // 0 | 1

    const int m0 = blockIdx.y * TM;
    const int n0 = blockIdx.x * TN;
    const int bz = blockIdx.z;

    A  += (size_t)bz * (size_t)aBatch;
    Bm += (size_t)bz * (size_t)bBatch;
    C  += (size_t)bz * (size_t)cBatch;

    float aF[16], bF[16];   // register-staged next tile (f32)

    // ---- stage A tile -> regs (uniform fast/slow branch, slow is branchless)
    auto stageA = [&](int k0) {
        if ((m0 + TM <= M) & (k0 + TK <= K) & ((lda & 3) == 0)) {
#pragma unroll
            for (int q = 0; q < 4; ++q) {
                int f = tid + (q << 8);
                int r = f >> 3, kq = (f & 7) << 2;
                float4 t = *(const float4*)(A + (size_t)(m0 + r) * lda + k0 + kq);
                aF[q*4+0] = t.x; aF[q*4+1] = t.y; aF[q*4+2] = t.z; aF[q*4+3] = t.w;
            }
        } else {
#pragma unroll
            for (int q = 0; q < 4; ++q) {
                int f = tid + (q << 8);
                int r = f >> 3, kq = (f & 7) << 2;
                int gm = m0 + r;
                int cm = gm < M ? gm : M - 1;
#pragma unroll
                for (int i = 0; i < 4; ++i) {
                    int gk = k0 + kq + i;
                    int ck = gk < K ? gk : K - 1;
                    float v = A[(size_t)cm * lda + ck];      // always issued (clamped)
                    aF[q*4+i] = (gm < M && gk < K) ? v : 0.0f;
                }
            }
        }
    };

    auto stageB = [&](int k0) {
        if (bTransposed) {
            if ((n0 + TN <= N) & (k0 + TK <= K) & ((ldb & 3) == 0)) {
#pragma unroll
                for (int q = 0; q < 4; ++q) {
                    int f = tid + (q << 8);
                    int kk = f >> 5, nq = (f & 31) << 2;
                    float4 t = *(const float4*)(Bm + (size_t)(k0 + kk) * ldb + n0 + nq);
                    bF[q*4+0] = t.x; bF[q*4+1] = t.y; bF[q*4+2] = t.z; bF[q*4+3] = t.w;
                }
            } else {
#pragma unroll
                for (int q = 0; q < 4; ++q) {
                    int f = tid + (q << 8);
                    int kk = f >> 5, nq = (f & 31) << 2;
                    int gk = k0 + kk;
                    int ck = gk < K ? gk : K - 1;
#pragma unroll
                    for (int i = 0; i < 4; ++i) {
                        int gn = n0 + nq + i;
                        int cn = gn < N ? gn : N - 1;
                        float v = Bm[(size_t)ck * ldb + cn];
                        bF[q*4+i] = (gk < K && gn < N) ? v : 0.0f;
                    }
                }
            }
        } else {
            if ((n0 + TN <= N) & (k0 + TK <= K) & ((ldb & 3) == 0)) {
#pragma unroll
                for (int q = 0; q < 4; ++q) {
                    int f = tid + (q << 8);
                    int r = f >> 3, kq = (f & 7) << 2;
                    float4 t = *(const float4*)(Bm + (size_t)(n0 + r) * ldb + k0 + kq);
                    bF[q*4+0] = t.x; bF[q*4+1] = t.y; bF[q*4+2] = t.z; bF[q*4+3] = t.w;
                }
            } else {
#pragma unroll
                for (int q = 0; q < 4; ++q) {
                    int f = tid + (q << 8);
                    int r = f >> 3, kq = (f & 7) << 2;
                    int gn = n0 + r;
                    int cn = gn < N ? gn : N - 1;
#pragma unroll
                    for (int i = 0; i < 4; ++i) {
                        int gk = k0 + kq + i;
                        int ck = gk < K ? gk : K - 1;
                        float v = Bm[(size_t)cn * ldb + ck];
                        bF[q*4+i] = (gn < N && gk < K) ? v : 0.0f;
                    }
                }
            }
        }
    };

    // ---- regs -> LDS (bf16), waits on the outstanding global loads
    auto writeA = [&]() {
#pragma unroll
        for (int q = 0; q < 4; ++q) {
            int f = tid + (q << 8);
            int r = f >> 3, kq = (f & 7) << 2;
            uint2 p;
            p.x = pack2bf(aF[q*4+0], aF[q*4+1]);
            p.y = pack2bf(aF[q*4+2], aF[q*4+3]);
            *(uint2*)(As + r * LDS_S + kq) = p;
        }
    };
    auto writeB = [&]() {
        if (bTransposed) {
#pragma unroll
            for (int q = 0; q < 4; ++q) {
                int f = tid + (q << 8);
                int kk = f >> 5, nq = (f & 31) << 2;
#pragma unroll
                for (int i = 0; i < 4; ++i)
                    Bs[(nq + i) * LDS_S + kk] = f2bf(bF[q*4+i]);
            }
        } else {
#pragma unroll
            for (int q = 0; q < 4; ++q) {
                int f = tid + (q << 8);
                int r = f >> 3, kq = (f & 7) << 2;
                uint2 p;
                p.x = pack2bf(bF[q*4+0], bF[q*4+1]);
                p.y = pack2bf(bF[q*4+2], bF[q*4+3]);
                *(uint2*)(Bs + r * LDS_S + kq) = p;
            }
        }
    };

    v8f acc[4][2];
    const v8f vzero = {0.f, 0.f, 0.f, 0.f, 0.f, 0.f, 0.f, 0.f};
#pragma unroll
    for (int i = 0; i < 4; ++i)
#pragma unroll
        for (int j = 0; j < 2; ++j) acc[i][j] = vzero;

    stageA(0);
    stageB(0);

    for (int k0 = 0; k0 < K; k0 += TK) {
        writeA();
        writeB();
        __syncthreads();

        // prefetch next tile into regs: global loads overlap ds reads + WMMAs
        if (k0 + TK < K) { stageA(k0 + TK); stageB(k0 + TK); }

        // ---- fragments (ISA 16-bit layouts) ----
        v16bf afrag[4], bfrag[2];
#pragma unroll
        for (int i = 0; i < 4; ++i) {
            int row = wm * 64 + i * 16 + laneRow;
            const v4u* p = (const v4u*)&As[row * LDS_S + (hiHalf << 3)];
            FragU fu; fu.u[0] = p[0]; fu.u[1] = p[2];   // +32B == k+16
            afrag[i] = fu.v;
        }
#pragma unroll
        for (int j = 0; j < 2; ++j) {
            int col = wn * 32 + j * 16 + laneRow;
            const v4u* p = (const v4u*)&Bs[col * LDS_S + (hiHalf << 3)];
            FragU fu; fu.u[0] = p[0]; fu.u[1] = p[2];
            bfrag[j] = fu.v;
        }

#pragma unroll
        for (int i = 0; i < 4; ++i)
#pragma unroll
            for (int j = 0; j < 2; ++j)
                acc[i][j] = __builtin_amdgcn_wmma_f32_16x16x32_bf16(
                    false, afrag[i], false, bfrag[j],
                    (short)0, acc[i][j], false, false);

        __syncthreads();
    }

    // ---- epilogue ----
#pragma unroll
    for (int i = 0; i < 4; ++i) {
        int mBase = m0 + wm * 64 + i * 16 + (hiHalf << 3);
#pragma unroll
        for (int j = 0; j < 2; ++j) {
            int n = n0 + wn * 32 + j * 16 + laneRow;
            if (n >= N) continue;
            float bv = bias ? bias[n] : 0.0f;
#pragma unroll
            for (int r = 0; r < 8; ++r) {
                int m = mBase + r;
                if (m < M) {
                    float v = acc[i][j][r] + bv;
                    if (mul) v *= mul[(size_t)m * ldmul + n];
                    C[(size_t)m * ldc + n] = v;
                }
            }
        }
    }
}

// ---------------------------------------------------------------------------
// Register-resident in-place row softmax (L <= 2048): one read + one write.
// ---------------------------------------------------------------------------
__global__ __launch_bounds__(256)
void hla_softmax_rows(float* __restrict__ X, int L)
{
    __shared__ float red[256];
    const int tid = threadIdx.x;
    float* x = X + (size_t)blockIdx.x * (size_t)L;

    float v[8];
    float m = -3.402823466e38f;
#pragma unroll
    for (int j = 0; j < 8; ++j) {
        int i = tid + (j << 8);
        v[j] = (i < L) ? x[i] : -3.402823466e38f;
        m = fmaxf(m, v[j]);
    }
    red[tid] = m; __syncthreads();
    for (int s = 128; s > 0; s >>= 1) {
        if (tid < s) red[tid] = fmaxf(red[tid], red[tid + s]);
        __syncthreads();
    }
    m = red[0]; __syncthreads();

    float sum = 0.0f;
#pragma unroll
    for (int j = 0; j < 8; ++j) {       // exp(-huge) == 0 for the pad lanes
        v[j] = __expf(v[j] - m);
        sum += v[j];
    }
    red[tid] = sum; __syncthreads();
    for (int s = 128; s > 0; s >>= 1) {
        if (tid < s) red[tid] += red[tid + s];
        __syncthreads();
    }
    float inv = 1.0f / red[0];
#pragma unroll
    for (int j = 0; j < 8; ++j) {
        int i = tid + (j << 8);
        if (i < L) x[i] = v[j] * inv;
    }
}

// ---------------------------------------------------------------------------
// Orchestration
// ---------------------------------------------------------------------------
extern "C" void kernel_launch(void* const* d_in, const int* in_sizes, int n_in,
                              void* d_out, int out_size, void* d_ws, size_t ws_size,
                              hipStream_t stream)
{
    (void)in_sizes; (void)n_in; (void)out_size; (void)ws_size;

    const float* H    = (const float*)d_in[0];   // [B, D, L]
    const float* Q1w  = (const float*)d_in[1];   // [N1, E]
    const float* Q2w  = (const float*)d_in[2];   // [N2, E]
    const float* Q12w = (const float*)d_in[3];   // [N2, N1]
    const float* Wmap = (const float*)d_in[4];   // [D, E]
    const float* bmap = (const float*)d_in[5];   // [D]

    constexpr int B = 8, D = 512, L = 2000, E = 300, N1 = 118, N2 = 8929;

    float* C2 = (float*)d_out;                       // [B, N2, D]
    float* A2 = C2 + (size_t)B * N2 * D;             // [B, N2, L]

    float* ws  = (float*)d_ws;
    float* Qm1 = ws;  ws += (size_t)N1 * D;          // [N1, D]
    float* Qm2 = ws;  ws += (size_t)N2 * D;          // [N2, D]
    float* E1  = ws;  ws += (size_t)B * N1 * L;      // [B, N1, L]  (softmaxed in place)
    float* C1  = ws;  ws += (size_t)B * N1 * D;      // [B, N1, D]
    float* Q2  = ws;  ws += (size_t)B * N2 * D;      // [B, N2, D]

    dim3 blk(256);
    auto g = [](int n, int t) { return (n + t - 1) / t; };

    // 1) Qm1 = Q1w @ Wmap^T + bmap   (Bt = Wmap[D,E], direct)
    hla_wmma_gemm<<<dim3(g(D,TN), g(N1,TM), 1), blk, 0, stream>>>(
        Q1w, Wmap, Qm1, N1, D, E, E, E, D, 0, 0, 0, bmap, nullptr, 0, 0);

    // 2) Qm2 = Q2w @ Wmap^T + bmap
    hla_wmma_gemm<<<dim3(g(D,TN), g(N2,TM), 1), blk, 0, stream>>>(
        Q2w, Wmap, Qm2, N2, D, E, E, E, D, 0, 0, 0, bmap, nullptr, 0, 0);

    // 3) E1[b] = Qm1 @ H[b]          (B = H[b] [D,L], transposed staging)
    hla_wmma_gemm<<<dim3(g(L,TN), g(N1,TM), B), blk, 0, stream>>>(
        Qm1, H, E1, N1, L, D, D, L, L,
        0, (long)D * L, (long)N1 * L, nullptr, nullptr, 0, 1);

    // 4) A1 = softmax(E1) over L, in place
    hla_softmax_rows<<<dim3(B * N1), blk, 0, stream>>>(E1, L);

    // 5) C1[b] = A1[b] @ H[b]^T      (Bt = H[b] [D,L], direct)
    hla_wmma_gemm<<<dim3(g(D,TN), g(N1,TM), B), blk, 0, stream>>>(
        E1, H, C1, N1, D, L, L, L, D,
        (long)N1 * L, (long)D * L, (long)N1 * D, nullptr, nullptr, 0, 0);

    // 6) Q2[b] = (Q12w @ C1[b]) * Qm2   (B = C1[b] [N1,D], transposed; mul=Qm2)
    hla_wmma_gemm<<<dim3(g(D,TN), g(N2,TM), B), blk, 0, stream>>>(
        Q12w, C1, Q2, N2, D, N1, N1, D, D,
        0, (long)N1 * D, (long)N2 * D, nullptr, Qm2, D, 1);

    // 7) E2[b] = Q2[b] @ H[b]  -> written directly to A2 output region
    hla_wmma_gemm<<<dim3(g(L,TN), g(N2,TM), B), blk, 0, stream>>>(
        Q2, H, A2, N2, L, D, D, L, L,
        (long)N2 * D, (long)D * L, (long)N2 * L, nullptr, nullptr, 0, 1);

    // 8) A2 = softmax(E2) over L, in place in the output buffer
    hla_softmax_rows<<<dim3(B * N2), blk, 0, stream>>>(A2, L);

    // 9) C2[b] = A2[b] @ H[b]^T      (Bt = H[b], direct)
    hla_wmma_gemm<<<dim3(g(D,TN), g(N2,TM), B), blk, 0, stream>>>(
        A2, H, C2, N2, D, L, L, L, D,
        (long)N2 * L, (long)D * L, (long)N2 * D, nullptr, nullptr, 0, 0);
}
